// TransformerBlock_41068477284968
// MI455X (gfx1250) — matrix operand order repaired and checked by reading, not compiled, not run on hardware
//
#include <hip/hip_runtime.h>
#include <math.h>

// ---------------------------------------------------------------------------
// TransformerBlock for MI455X (gfx1250): 5-pass WMMA f16 pipeline.
//   pass A: S   = (Q @ K^T) * 1/(sqrt(D)+eps)          [wmma f16 -> f32]
//   pass B: P   = masked softmax(S)                    [VALU + LDS reduce]
//   pass C: X   = Q + P @ V                            [wmma]
//   pass D: h   = relu(X @ W1^T + b1)                  [wmma]
//   pass E: out = h @ W2^T + b2 + X                    [wmma]
// GEMM staging uses CDNA5 async copy: global_load_async_to_lds_b128 +
// s_wait_asynccnt, with explicit LDS double buffering.
// ---------------------------------------------------------------------------

typedef _Float16 half_t;
typedef __attribute__((ext_vector_type(16))) _Float16 v16h;
typedef __attribute__((ext_vector_type(8)))  float    v8f;

#define B_  32
#define LQ_ 1024
#define LK_ 1024
#define D_  1024

// ----------------------------- async copy helpers ---------------------------

// Copy 16 bytes from global to LDS via the async copy engine (ASYNCcnt).
// VDST = VGPR holding LDS byte address, VADDR = 64-bit global address.
__device__ __forceinline__ void async_cp16(unsigned lds_addr, const void* gptr) {
  asm volatile("global_load_async_to_lds_b128 %0, %1, off"
               :: "v"(lds_addr), "v"((unsigned long long)(size_t)gptr)
               : "memory");
}

__device__ __forceinline__ void wait_async0() {
  asm volatile("s_wait_asynccnt 0x0" ::: "memory");
}

// ----------------------------- converts ------------------------------------

__global__ __launch_bounds__(256) void cvt_f32_f16(const float* __restrict__ src,
                                                   half_t* __restrict__ dst, long n) {
  long i = (long)blockIdx.x * blockDim.x + threadIdx.x;
  long stride = (long)gridDim.x * blockDim.x;
  for (; i < n; i += stride) dst[i] = (half_t)src[i];
}

// Vt[b][d][k] = (f16) V[b][k][d]  (so pass C's B-operand is row-major-NT)
__global__ __launch_bounds__(256) void transpose_cvt(const float* __restrict__ V,
                                                     half_t* __restrict__ Vt) {
  __shared__ float tile[32][33];
  const int b = blockIdx.z;
  const int k0 = blockIdx.x * 32, d0 = blockIdx.y * 32;
  const float* Vb = V + (long)b * LK_ * D_;
  half_t* Vtb = Vt + (long)b * D_ * LK_;
  const int tx = threadIdx.x & 31, ty = threadIdx.x >> 5;  // 32 x 8
  for (int r = ty; r < 32; r += 8)
    tile[r][tx] = Vb[(long)(k0 + r) * D_ + d0 + tx];
  __syncthreads();
  for (int r = ty; r < 32; r += 8)
    Vtb[(long)(d0 + r) * LK_ + k0 + tx] = (half_t)tile[tx][r];
}

// ----------------------------- softmax -------------------------------------
// p_k = m_k * exp(s_k - max_unmasked) / (sum_j m_j exp(s_j - max) + 1e-13)
__global__ __launch_bounds__(256) void softmax_mask(const float* __restrict__ S,
                                                    const float* __restrict__ Vmask,
                                                    half_t* __restrict__ P) {
  const int q = blockIdx.x, b = blockIdx.y;
  const float* s = S + ((long)b * LQ_ + q) * LK_;
  const float* m = Vmask + (long)b * LK_;
  half_t* p = P + ((long)b * LQ_ + q) * LK_;
  __shared__ float red[256];
  const int t = threadIdx.x;
  float sv[4], mv[4];
  float lmax = -3.0e38f;
  for (int i = 0; i < 4; i++) {
    int c = t + i * 256;
    sv[i] = s[c];
    mv[i] = m[c];
    if (mv[i] > 0.f && sv[i] > lmax) lmax = sv[i];
  }
  red[t] = lmax; __syncthreads();
  for (int w = 128; w > 0; w >>= 1) {
    if (t < w) red[t] = fmaxf(red[t], red[t + w]);
    __syncthreads();
  }
  float mx = red[0];
  if (mx < -1e37f) mx = 0.f;  // fully masked row -> all outputs 0 anyway
  __syncthreads();
  float e[4], lsum = 0.f;
  for (int i = 0; i < 4; i++) {
    e[i] = (mv[i] > 0.f) ? __expf(sv[i] - mx) : 0.f;
    lsum += e[i];
  }
  red[t] = lsum; __syncthreads();
  for (int w = 128; w > 0; w >>= 1) {
    if (t < w) red[t] += red[t + w];
    __syncthreads();
  }
  const float inv = 1.f / (red[0] + 1e-13f);
  for (int i = 0; i < 4; i++) {
    int c = t + i * 256;
    p[c] = (half_t)(e[i] * inv);
  }
}

// ----------------------------- WMMA NT GEMM --------------------------------
// C[M,N] = A[M,Kd] @ B[N,Kd]^T.  Workgroup: 256 thr (8 waves), tile 128x128.
// Wave (4x2 grid) owns 32x64 -> 2x4 accumulators of v_wmma_f32_16x16x32_f16.
// Async-copy double-buffered LDS staging.
// MODE 0: C=acc*scale (f32)              [scores]
// MODE 1: x=E+acc; C=x (f32); Ch=x(f16)  [attention out + residual]
// MODE 2: Ch=relu(acc+bias[col]) (f16)   [FFN layer 1]
// MODE 3: C=acc+bias[col]+E (f32)        [FFN layer 2 + residual]
template <int MODE>
__global__ __launch_bounds__(256) void gemm_nt(
    const half_t* __restrict__ A, long strideA,
    const half_t* __restrict__ Bm, long strideB,
    float* __restrict__ C, long strideC,
    half_t* __restrict__ Ch, long strideCh,
    const float* __restrict__ E, long strideE,
    const float* __restrict__ bias,
    int M, int N, int Kd, float scale) {
  // 40-element rows (pad 8) -> conflict-free 16B ds loads; 2 buffers = 40 KB
  __shared__ __align__(16) half_t As[2][128][40];
  __shared__ __align__(16) half_t Bs[2][128][40];

  const int bz = blockIdx.z;
  A += (long)bz * strideA;
  Bm += (long)bz * strideB;
  if (MODE == 0 || MODE == 1 || MODE == 3) C += (long)bz * strideC;
  if (MODE == 1 || MODE == 2) Ch += (long)bz * strideCh;
  if (MODE == 1 || MODE == 3) E += (long)bz * strideE;

  const int tileM = blockIdx.y * 128, tileN = blockIdx.x * 128;
  const int t = threadIdx.x;
  const int wave = t >> 5, lane = t & 31;
  const int waveM = wave >> 1, waveN = wave & 1;  // 4 x 2 wave grid
  const int lhalf = lane >> 4, lm = lane & 15;

  union Frag { v16h v; uint4 q[2]; };
  v8f acc[2][4];
  for (int i = 0; i < 2; i++)
    for (int j = 0; j < 4; j++)
      for (int k = 0; k < 8; k++) acc[i][j][k] = 0.f;

  // staging: 2 threads per tile row, 32B (two b128 async copies) each
  const int r = t >> 1, seg = t & 1;
  const half_t* Arow = A + (long)(tileM + r) * Kd + seg * 16;
  const half_t* Brow = Bm + (long)(tileN + r) * Kd + seg * 16;
  const unsigned ldsA0 = (unsigned)(size_t)&As[0][r][seg * 16];
  const unsigned ldsB0 = (unsigned)(size_t)&Bs[0][r][seg * 16];
  const unsigned bufA = (unsigned)sizeof(As[0]);  // 10240 B
  const unsigned bufB = (unsigned)sizeof(Bs[0]);

  // prologue: prefetch tile 0 into buffer 0
  async_cp16(ldsA0, Arow);
  async_cp16(ldsA0 + 16, Arow + 8);
  async_cp16(ldsB0, Brow);
  async_cp16(ldsB0 + 16, Brow + 8);

  int buf = 0;
  for (int k0 = 0; k0 < Kd; k0 += 32, buf ^= 1) {
    wait_async0();       // this wave's async writes into `buf` have landed
    __syncthreads();     // ... and everyone else's too; `buf^1` fully consumed

    if (k0 + 32 < Kd) {  // prefetch next tile into the other buffer
      const unsigned oA = ldsA0 + (buf ^ 1) * bufA;
      const unsigned oB = ldsB0 + (buf ^ 1) * bufB;
      async_cp16(oA, Arow + k0 + 32);
      async_cp16(oA + 16, Arow + k0 + 40);
      async_cp16(oB, Brow + k0 + 32);
      async_cp16(oB + 16, Brow + k0 + 40);
    }

    // A fragment (16x32, CDNA5 layout): lane half -> K {0..7,16..23}/{8..15,24..31}
    Frag af[2], bf[4];
    for (int mi = 0; mi < 2; mi++) {
      const int row = waveM * 32 + mi * 16 + lm;
      af[mi].q[0] = *(const uint4*)&As[buf][row][8 * lhalf];
      af[mi].q[1] = *(const uint4*)&As[buf][row][16 + 8 * lhalf];
    }
    // B fragment (32x16): lane = column, lane half -> K {0..7,16..23}/{8..15,24..31}
    for (int ni = 0; ni < 4; ni++) {
      const int col = waveN * 64 + ni * 16 + lm;
      bf[ni].q[0] = *(const uint4*)&Bs[buf][col][8 * lhalf];
      bf[ni].q[1] = *(const uint4*)&Bs[buf][col][16 + 8 * lhalf];
    }
    for (int mi = 0; mi < 2; mi++)
      for (int ni = 0; ni < 4; ni++)
        acc[mi][ni] = __builtin_amdgcn_wmma_f32_16x16x32_f16(
            false, af[mi].v, false, bf[ni].v, (short)0, acc[mi][ni], false, false);
  }

  // epilogue: C/D layout: vgpr rr, lane -> (row = rr + 8*lhalf, col = lm)
  for (int mi = 0; mi < 2; mi++)
    for (int ni = 0; ni < 4; ni++)
      for (int rr = 0; rr < 8; rr++) {
        const int gr = tileM + waveM * 32 + mi * 16 + rr + 8 * lhalf;
        const int gc = tileN + waveN * 64 + ni * 16 + lm;
        const float v = acc[mi][ni][rr];
        const long idx = (long)gr * N + gc;
        if (MODE == 0) {
          C[idx] = v * scale;
        } else if (MODE == 1) {
          const float x = E[idx] + v;
          C[idx] = x;
          Ch[idx] = (half_t)x;
        } else if (MODE == 2) {
          const float h = v + bias[gc];
          Ch[idx] = (half_t)(h > 0.f ? h : 0.f);
        } else {
          C[idx] = v + bias[gc] + E[idx];
        }
      }
}

// ----------------------------- launcher ------------------------------------

extern "C" void kernel_launch(void* const* d_in, const int* in_sizes, int n_in,
                              void* d_out, int out_size, void* d_ws, size_t ws_size,
                              hipStream_t stream) {
  const float* Q  = (const float*)d_in[0];
  const float* K  = (const float*)d_in[1];
  const float* V  = (const float*)d_in[2];
  const float* Vm = (const float*)d_in[3];
  const float* W1 = (const float*)d_in[4];
  const float* b1 = (const float*)d_in[5];
  const float* W2 = (const float*)d_in[6];
  const float* b2 = (const float*)d_in[7];
  float* Out = (float*)d_out;

  // workspace layout (regions reused as tensors die):
  //  R0 [0,      64MB): Qh  -> P
  //  R1 [64MB,  128MB): Kh  -> Xb
  //  R2 [128MB, 192MB): Vt  -> h
  //  R3 [192MB, 320MB): S(f32) -> Xf(f32)
  //  R4 [320MB, 324MB): W1h, W2h
  char* ws = (char*)d_ws;
  const long NB = (long)B_ * LQ_ * D_;  // 33,554,432 elements
  half_t* Qh  = (half_t*)(ws);
  half_t* Kh  = (half_t*)(ws + NB * 2);
  half_t* Vt  = (half_t*)(ws + NB * 4);
  float*  S   = (float*)(ws + NB * 6);
  half_t* W1h = (half_t*)(ws + NB * 6 + NB * 4);
  half_t* W2h = (half_t*)(ws + NB * 6 + NB * 4 + (long)D_ * D_ * 2);
  half_t* P = Qh;   // reuse
  half_t* Xb = Kh;  // reuse
  half_t* Hh = Vt;  // reuse
  float*  Xf = S;   // reuse

  // 1) convert operands to f16 (V transposed to [b][d][k])
  cvt_f32_f16<<<4096, 256, 0, stream>>>(Q, Qh, NB);
  cvt_f32_f16<<<4096, 256, 0, stream>>>(K, Kh, NB);
  cvt_f32_f16<<<512, 256, 0, stream>>>(W1, W1h, (long)D_ * D_);
  cvt_f32_f16<<<512, 256, 0, stream>>>(W2, W2h, (long)D_ * D_);
  transpose_cvt<<<dim3(LK_ / 32, D_ / 32, B_), 256, 0, stream>>>(V, Vt);

  const float scale = 1.0f / (sqrtf((float)D_) + 1e-8f);

  // 2) S = Q @ K^T * scale     (batched)
  gemm_nt<0><<<dim3(LK_ / 128, LQ_ / 128, B_), 256, 0, stream>>>(
      Qh, (long)LQ_ * D_, Kh, (long)LK_ * D_, S, (long)LQ_ * LK_,
      nullptr, 0, nullptr, 0, nullptr, LQ_, LK_, D_, scale);

  // 3) P = masked softmax(S)
  softmax_mask<<<dim3(LQ_, B_), 256, 0, stream>>>(S, Vm, P);

  // 4) X = Q + P @ V           (batched; Vt is V^T per batch)
  gemm_nt<1><<<dim3(D_ / 128, LQ_ / 128, B_), 256, 0, stream>>>(
      P, (long)LQ_ * LK_, Vt, (long)D_ * LK_, Xf, (long)LQ_ * D_,
      Xb, (long)LQ_ * D_, Q, (long)LQ_ * D_, nullptr, LQ_, D_, LK_, 1.f);

  // 5) h = relu(X @ W1^T + b1) (batch folded into M)
  gemm_nt<2><<<dim3(D_ / 128, (B_ * LQ_) / 128, 1), 256, 0, stream>>>(
      Xb, 0, W1h, 0, nullptr, 0, Hh, 0, nullptr, 0, b1,
      B_ * LQ_, D_, D_, 1.f);

  // 6) out = h @ W2^T + b2 + X
  gemm_nt<3><<<dim3(D_ / 128, (B_ * LQ_) / 128, 1), 256, 0, stream>>>(
      Hh, 0, W2h, 0, Out, 0, nullptr, 0, Xf, 0, b2,
      B_ * LQ_, D_, D_, 1.f);
}
